// RNN_62268435857828
// MI455X (gfx1250) — matrix operand order repaired and checked by reading, not compiled, not run on hardware
//
#include <hip/hip_runtime.h>
#include <hip/hip_bf16.h>

// Problem constants (from reference)
#define B_BATCH 2048
#define T_SEQ   512
#define EMBED   64
#define HIDDEN  100
#define HP      112   // HIDDEN padded to 7*16 (WMMA N tiles)
#define KP      128   // K (hidden) padded to 4*32 (WMMA bf16 K tiles)
#define NT      7     // number of 16-wide N tiles
#define HB_STRIDE 136 // padded LDS row stride (bf16 elems): 272B rows -> 16B aligned
                      // chunks, bank-conflict-free row accesses

typedef __attribute__((ext_vector_type(16))) __bf16 v16bf;
typedef __attribute__((ext_vector_type(8)))  __bf16 v8bf;
typedef __attribute__((ext_vector_type(8)))  float  v8f;

static __device__ __forceinline__ v8f wmma_bf16(v16bf a, v16bf b, v8f c) {
  // D = A(16x32 bf16) * B(32x16 bf16) + C(16x16 f32)
  return __builtin_amdgcn_wmma_f32_16x16x32_bf16(
      /*neg_a=*/false, a, /*neg_b=*/false, b,
      /*c_mod=*/(short)0, c, /*reuse_a=*/false, /*reuse_b=*/false);
}

// Async copy 16B global -> LDS, tracked by ASYNCcnt (gfx1250).
static __device__ __forceinline__ void async_cp_b128(unsigned lds_addr,
                                                     const void* gaddr) {
  asm volatile("global_load_async_to_lds_b128 %0, %1, off"
               :: "v"(lds_addr), "v"(gaddr) : "memory");
}
static __device__ __forceinline__ void wait_asynccnt0() {
  asm volatile("s_wait_asynccnt 0" ::: "memory");
}

// ---------------------------------------------------------------------------
// Kernel 0: repack weights into WMMA-friendly padded bf16 B-matrix layouts.
//   WihB[k][n] = W_ih[n][k]   (k<64,  n<112; zero-pad n>=100)
//   WhhB[k][n] = W_hh[n][k]   (k<128, n<112; zero-pad k>=100 or n>=100)
//   bb[n]      = b_ih[n] + b_hh[n] (zero-pad)
// ---------------------------------------------------------------------------
__global__ void __launch_bounds__(256) prep_kernel(
    const float* __restrict__ W_ih, const float* __restrict__ W_hh,
    const float* __restrict__ b_ih, const float* __restrict__ b_hh,
    __bf16* __restrict__ WihB, __bf16* __restrict__ WhhB,
    float* __restrict__ bb) {
  int tid = blockIdx.x * blockDim.x + threadIdx.x;
  int nth = gridDim.x * blockDim.x;
  for (int i = tid; i < EMBED * HP; i += nth) {
    int k = i / HP, n = i % HP;
    float v = (n < HIDDEN) ? W_ih[n * EMBED + k] : 0.0f;
    WihB[i] = (__bf16)v;
  }
  for (int i = tid; i < KP * HP; i += nth) {
    int k = i / HP, n = i % HP;
    float v = (k < HIDDEN && n < HIDDEN) ? W_hh[n * HIDDEN + k] : 0.0f;
    WhhB[i] = (__bf16)v;
  }
  for (int i = tid; i < HP; i += nth)
    bb[i] = (i < HIDDEN) ? (b_ih[i] + b_hh[i]) : 0.0f;
}

// ---------------------------------------------------------------------------
// Kernel 1: xproj = embed_gather(x) @ W_ih^T + (b_ih + b_hh), stored bf16.
// One wave per 16x112 output tile (16 batch rows at one timestep), grid-stride
// over tiles. W_ih B-fragments + bias hoisted into registers outside the loop.
// Layout: xp[bt][t][16][HP], tile index = bt*T + t (what the scan consumes).
// Embedding table (~12.8 MB) is L2-resident, so gathers are cheap; this kernel
// is bounded by the bf16 xproj store stream (~235 MB).
// ---------------------------------------------------------------------------
__global__ void __launch_bounds__(256) xproj_kernel(
    const int* __restrict__ x,            // [B][T] token ids
    const float* __restrict__ embed,      // [VOCAB][64] f32
    const __bf16* __restrict__ WihB,      // [64][112] bf16
    const float* __restrict__ bb,         // [112]
    __bf16* __restrict__ xp)              // [(B/16)*T][16][112] bf16
{
  __shared__ __bf16 wih[EMBED * HP];
  __shared__ float  bbs[HP];
  int tid = threadIdx.x;
  for (int i = tid; i < EMBED * HP; i += blockDim.x) wih[i] = WihB[i];
  for (int i = tid; i < HP; i += blockDim.x) bbs[i] = bb[i];
  __syncthreads();

  int lane = tid & 31;
  int wave = tid >> 5;
  int half = lane >> 4;   // which 16-lane half
  int m    = lane & 15;   // A-matrix row / C-matrix column index

  // B fragments of W_ih: 7 n-tiles x 2 k-tiles, kept in VGPRs.
  // B layout (16-bit, 32x16), same K order as A: vector elem e -> K =
  // e<8 ? half*8 + e : 16 + half*8 + (e-8).
  v16bf bfrag[NT][2];
  float bias[NT];
#pragma unroll
  for (int nt = 0; nt < NT; ++nt) {
    int n0 = nt * 16;
#pragma unroll
    for (int kt = 0; kt < 2; ++kt)
#pragma unroll
      for (int e = 0; e < 16; ++e)
        bfrag[nt][kt][e] = wih[(kt * 32 + ((e < 8) ? (half * 8 + e) : (16 + half * 8 + (e - 8)))) * HP + n0 + m];
    bias[nt] = bbs[n0 + m];
  }

  const int nTiles = (B_BATCH / 16) * T_SEQ;      // 65536
  int wavesTotal = gridDim.x * (blockDim.x >> 5);
  for (int tile = blockIdx.x * (blockDim.x >> 5) + wave; tile < nTiles;
       tile += wavesTotal) {
    int bt = tile / T_SEQ;
    int t  = tile % T_SEQ;
    int b  = bt * 16 + m;
    int tok = x[b * T_SEQ + t];
    const float* erow = embed + (size_t)tok * EMBED;

    // A fragments (16x32 bf16 per k-tile). Vector elem e -> K offset:
    //   e<8 : half*8 + e          (VGPR 0-3 region)
    //   e>=8: 16 + half*8 + (e-8) (VGPR 4-7 region)
    v16bf a[2];
#pragma unroll
    for (int kt = 0; kt < 2; ++kt)
#pragma unroll
      for (int e = 0; e < 16; ++e) {
        int k = kt * 32 + ((e < 8) ? (half * 8 + e) : (16 + half * 8 + (e - 8)));
        a[kt][e] = (__bf16)erow[k];
      }

    __bf16* orow = xp + (size_t)tile * 16 * HP;
#pragma unroll
    for (int nt = 0; nt < NT; ++nt) {
      v8f c;
#pragma unroll
      for (int r = 0; r < 8; ++r) c[r] = bias[nt];
      c = wmma_bf16(a[0], bfrag[nt][0], c);
      c = wmma_bf16(a[1], bfrag[nt][1], c);
      // C/D layout: VGPR r holds (M = r + 8*half, N = lane&15)
#pragma unroll
      for (int r = 0; r < 8; ++r)
        orow[(r + 8 * half) * HP + nt * 16 + m] = (__bf16)c[r];
    }
  }
}

// ---------------------------------------------------------------------------
// Kernel 2: the recurrence. One block per 16-row batch tile; 7 waves, each
// owning one 16-wide slice of the hidden state. h_t exchanged via LDS (bf16,
// padded stride) with two workgroup barriers per step. W_hh B-fragments in
// registers. The xproj tile for step t+1 is double-buffered in LDS via
// GLOBAL_LOAD_ASYNC_TO_LDS_B128 (224 threads x 16B = one 3584B tile), so the
// HBM stream never sits on the recurrence critical path.
// ---------------------------------------------------------------------------
__global__ void __launch_bounds__(224) rnn_scan_kernel(
    const __bf16* __restrict__ xp,     // [(B/16)*T][16][112] bf16
    const __bf16* __restrict__ WhhB,   // [128][112] bf16
    const float* __restrict__ fc_W,    // [100]
    const float* __restrict__ fc_b,    // [1]
    float* __restrict__ out)           // [2048]
{
  __shared__ alignas(16) __bf16 whh[KP * HP];          // 28 KB
  __shared__ alignas(16) __bf16 hbf[16 * HB_STRIDE];   // h_t bf16, padded rows
  __shared__ alignas(16) __bf16 xtile[2][16 * HP];     // async double buffer
  __shared__ float  hf[16 * HP];                       // final-step h in f32

  int tid  = threadIdx.x;
  int lane = tid & 31;
  int wave = tid >> 5;        // 0..6 -> n-tile
  int half = lane >> 4;
  int m    = lane & 15;
  int bt   = blockIdx.x;      // batch tile 0..127
  int n0   = wave * 16;

  const __bf16* xpb = xp + (size_t)bt * T_SEQ * 16 * HP;
  // Per-thread async-copy slot: thread tid moves bytes [tid*16, tid*16+16).
  unsigned lds_dst[2];
  lds_dst[0] = (unsigned)(uintptr_t)(&xtile[0][0]) + (unsigned)tid * 16u;
  lds_dst[1] = (unsigned)(uintptr_t)(&xtile[1][0]) + (unsigned)tid * 16u;

  // Stage tile t=0 while we also stage W_hh and zero h.
  async_cp_b128(lds_dst[0], xpb + tid * 8);

  for (int i = tid; i < KP * HP; i += blockDim.x) whh[i] = WhhB[i];
  for (int i = tid; i < 16 * HB_STRIDE; i += blockDim.x) hbf[i] = (__bf16)0.0f;
  wait_asynccnt0();
  __syncthreads();

  // W_hh B fragments for this wave's n-tile: 4 k-tiles, in registers.
  v16bf bfrag[4];
#pragma unroll
  for (int kt = 0; kt < 4; ++kt)
#pragma unroll
    for (int e = 0; e < 16; ++e)
      bfrag[kt][e] = whh[(kt * 32 + ((e < 8) ? (half * 8 + e) : (16 + half * 8 + (e - 8)))) * HP + n0 + m];

  for (int t = 0; t < T_SEQ; ++t) {
    // Kick the async stage of tile t+1 into the other buffer. Its previous
    // contents (tile t-1) were fully consumed before the last barrier.
    if (t + 1 < T_SEQ)
      async_cp_b128(lds_dst[(t + 1) & 1],
                    xpb + (size_t)(t + 1) * 16 * HP + tid * 8);

    // A fragments: h_{t-1} rows from LDS. Each k-tile is two contiguous
    // 8-elem (16B, aligned) bf16 chunks -> ds_load_b128 pairs.
    const __bf16* hrow = &hbf[m * HB_STRIDE];
    v16bf a[4];
#pragma unroll
    for (int kt = 0; kt < 4; ++kt) {
      v8bf lo = *(const v8bf*)(hrow + kt * 32 + half * 8);
      v8bf hi = *(const v8bf*)(hrow + kt * 32 + 16 + half * 8);
      a[kt] = __builtin_shufflevector(lo, hi, 0, 1, 2, 3, 4, 5, 6, 7,
                                      8, 9, 10, 11, 12, 13, 14, 15);
    }

    // C init from the LDS-staged xproj tile (already contains b_ih + b_hh).
    const __bf16* xt = &xtile[t & 1][0];
    v8f c;
#pragma unroll
    for (int r = 0; r < 8; ++r)
      c[r] = (float)xt[(r + 8 * half) * HP + n0 + m];

#pragma unroll
    for (int kt = 0; kt < 4; ++kt)
      c = wmma_bf16(a[kt], bfrag[kt], c);

    float hv[8];
#pragma unroll
    for (int r = 0; r < 8; ++r) hv[r] = tanhf(c[r]);

    __syncthreads();   // everyone done reading h_{t-1}
#pragma unroll
    for (int r = 0; r < 8; ++r)
      hbf[(r + 8 * half) * HB_STRIDE + n0 + m] = (__bf16)hv[r];
    if (t == T_SEQ - 1) {
#pragma unroll
      for (int r = 0; r < 8; ++r)
        hf[(r + 8 * half) * HP + n0 + m] = hv[r];
    }
    wait_asynccnt0();  // tile t+1 has landed in LDS
    __syncthreads();   // publish h_t and the staged tile
  }

  // Final projection + sigmoid: 16 lanes of wave 0, one batch row each.
  if (wave == 0 && lane < 16) {
    float acc = fc_b[0];
    for (int k = 0; k < HIDDEN; ++k)
      acc += hf[lane * HP + k] * fc_W[k];
    out[bt * 16 + lane] = 1.0f / (1.0f + __expf(-acc));
  }
}

// ---------------------------------------------------------------------------
extern "C" void kernel_launch(void* const* d_in, const int* in_sizes, int n_in,
                              void* d_out, int out_size, void* d_ws, size_t ws_size,
                              hipStream_t stream) {
  const int*   x      = (const int*)d_in[0];
  const float* embed  = (const float*)d_in[1];
  const float* W_ih   = (const float*)d_in[2];
  const float* W_hh   = (const float*)d_in[3];
  const float* b_ih   = (const float*)d_in[4];
  const float* b_hh   = (const float*)d_in[5];
  const float* fc_W   = (const float*)d_in[6];
  const float* fc_b   = (const float*)d_in[7];
  float* out = (float*)d_out;

  // Workspace layout (bytes):
  //   xp   : (B/16)*T*16*112 bf16  = ~235 MB
  //   WihB : 64*112 bf16
  //   WhhB : 128*112 bf16
  //   bb   : 112 f32
  char* ws = (char*)d_ws;
  size_t off = 0;
  __bf16* xp = (__bf16*)(ws + off);
  off += ((size_t)(B_BATCH / 16) * T_SEQ * 16 * HP * sizeof(__bf16) + 255) & ~(size_t)255;
  __bf16* WihB = (__bf16*)(ws + off);
  off += ((size_t)EMBED * HP * sizeof(__bf16) + 255) & ~(size_t)255;
  __bf16* WhhB = (__bf16*)(ws + off);
  off += ((size_t)KP * HP * sizeof(__bf16) + 255) & ~(size_t)255;
  float* bb = (float*)(ws + off);

  prep_kernel<<<1, 256, 0, stream>>>(W_ih, W_hh, b_ih, b_hh, WihB, WhhB, bb);
  xproj_kernel<<<1024, 256, 0, stream>>>(x, embed, WihB, bb, xp);
  rnn_scan_kernel<<<B_BATCH / 16, 224, 0, stream>>>(xp, WhhB, fc_W, fc_b, out);
}
